// SuperPointMatching_73289321939488
// MI455X (gfx1250) — matrix operand order repaired and checked by reading, not compiled, not run on hardware
//
#include <hip/hip_runtime.h>

typedef __attribute__((ext_vector_type(16))) _Float16 v16h;
typedef __attribute__((ext_vector_type(8)))  _Float16 v8h;
typedef __attribute__((ext_vector_type(8)))  float    v8f;
typedef __attribute__((ext_vector_type(4)))  int      v4i;

#define NPTS   8192
#define FDIM   256
#define KTOP   512
#define HBINS  4096
#define CAP    4096
#define BSTRIDE 72                        // 64 n + 8 pad f16 (144B rows, 16B aligned)

// workspace layout in dwords
#define WS_ROWSUM  0
#define WS_COLSUM  8192
#define WS_REFNN   16384
#define WS_SRCNN   24576
#define WS_HIST    32768
#define WS_CNT     36864
#define WS_THR     36865
#define WS_CAND    36872                 // uint2[CAP] = 8192 dwords (8B aligned)
#define WS_REF16   65536                 // 8192*256 f16 = 1048576 dwords
#define WS_SRCT16  (65536 + 1048576)     // transposed [256][8192] f16
#define WS_ZERO_DWORDS 45064

// ------------------------------------------------------- async-copy helpers

__device__ __forceinline__ void async_b128_to_lds(const _Float16* g, _Float16* l) {
#if __has_builtin(__builtin_amdgcn_global_load_async_to_lds_b128)
    // probe-observed signature: (int4* gaddr, int4* ldsaddr, imm offset, imm cpol)
    __builtin_amdgcn_global_load_async_to_lds_b128((v4i*)g, (v4i*)l, 0, 0);
#else
    unsigned laddr = (unsigned)(unsigned long long)
        (__attribute__((address_space(3))) void*)l;
    unsigned long long gaddr = (unsigned long long)g;
    asm volatile("global_load_async_to_lds_b128 %0, %1, off"
                 :: "v"(laddr), "v"(gaddr) : "memory");
#endif
}

__device__ __forceinline__ void wait_async0() {
#if __has_builtin(__builtin_amdgcn_s_wait_asynccnt)
    __builtin_amdgcn_s_wait_asynccnt(0);
#else
    asm volatile("s_wait_asynccnt 0x0" ::: "memory");
#endif
}

// ---------------------------------------------------------------- utilities

__global__ void init_ws_kernel(unsigned* ws) {
    int i = blockIdx.x * blockDim.x + threadIdx.x;
    if (i < WS_ZERO_DWORDS) ws[i] = 0u;
}

// fp32 -> f16, row-major copy (ref feats)
__global__ void cvt_f16_kernel(const float* __restrict__ in, _Float16* __restrict__ out) {
    int i = blockIdx.x * blockDim.x + threadIdx.x;
    out[i] = (_Float16)in[i];
}

// fp32 [8192][256] -> f16 transposed [256][8192] (src feats, K-major for B frags)
__global__ void cvt_f16_T_kernel(const float* __restrict__ in, _Float16* __restrict__ outT) {
    __shared__ float tile[16][17];
    int kb = blockIdx.x * 16;            // k tile base (0..255)
    int nb = blockIdx.y * 16;            // n tile base (0..8191)
    int tx = threadIdx.x & 15;
    int ty = threadIdx.x >> 4;
    tile[ty][tx] = in[(nb + ty) * FDIM + kb + tx];
    __syncthreads();
    outT[(size_t)(kb + ty) * NPTS + nb + tx] = (_Float16)tile[tx][ty];
}

// nearest-neighbor (exclude self: d < 1e-6  <=>  d2 < 1e-12)
__global__ void nn_kernel(const float* __restrict__ pts, int* __restrict__ nn) {
    __shared__ float sp[256 * 3];
    int i = blockIdx.x * 256 + threadIdx.x;
    float px = pts[i * 3 + 0], py = pts[i * 3 + 1], pz = pts[i * 3 + 2];
    float best = 3.4e38f;
    int bi = 0;
    for (int base = 0; base < NPTS; base += 256) {
        int t = threadIdx.x;
        sp[t]       = pts[base * 3 + t];
        sp[256 + t] = pts[base * 3 + 256 + t];
        sp[512 + t] = pts[base * 3 + 512 + t];
        __syncthreads();
        for (int j = 0; j < 256; ++j) {
            float dx = px - sp[j * 3 + 0];
            float dy = py - sp[j * 3 + 1];
            float dz = pz - sp[j * 3 + 2];
            float d2 = dx * dx + dy * dy + dz * dz;
            if (d2 >= 1e-12f && d2 < best) { best = d2; bi = base + j; }
        }
        __syncthreads();
    }
    nn[i] = bi;
}

// ------------------------------------------------- WMMA block tile 128x64
// Block = 8 waves; wave w computes m-tile (blockIdx.x*8+w) x 64-wide n strip.
// B chunk (32k x 64n) is staged once per block into LDS with async copies,
// double-buffered so the prefetch of k-step ks+1 overlaps the WMMAs of ks.
//
// A frag (ISA 16-bit A layout): lane m=l16, khalf=(lane>=16)?8:0,
//   elems 0..7 -> k = khalf+0..7 ; elems 8..15 -> k = 16+khalf+0..7
// B frag (ISA 16-bit B layout): lane n=l16, elems 0..7 -> k = khalf+0..7 ; elems 8..15 -> k = 16+khalf+0..7

__device__ __forceinline__ void stage_b_issue(const _Float16* __restrict__ BT16,
                                              _Float16* dst, int nbase, int ks) {
    int t   = threadIdx.x;
    int row = t >> 3;                    // 0..31 (k within chunk)
    int seg = t & 7;                     // 16B segment within 128B row
    const _Float16* g = BT16 + (size_t)(ks * 32 + row) * NPTS + nbase + seg * 8;
    async_b128_to_lds(g, dst + row * BSTRIDE + seg * 8);
}

__device__ __forceinline__ void wmma_strip(const _Float16* __restrict__ A16,
                                           const _Float16* __restrict__ BT16,
                                           _Float16* bstage,   // [2][32][BSTRIDE]
                                           int mtile, int nbase, v8f acc[4]) {
    const int lane  = threadIdx.x & 31;
    const int l16   = lane & 15;
    const int khalf = (lane >> 4) << 3;   // 0 or 8
    const _Float16* arow = A16 + (size_t)(mtile * 16 + l16) * FDIM;
#pragma unroll
    for (int t = 0; t < 4; ++t) acc[t] = (v8f){0.f,0.f,0.f,0.f,0.f,0.f,0.f,0.f};

    stage_b_issue(BT16, bstage, nbase, 0);           // prologue: fill buffer 0
#pragma unroll
    for (int ks = 0; ks < 8; ++ks) {
        wait_async0();                               // my chunk-ks copy done
        __syncthreads();                             // everyone's copy visible
        if (ks < 7)                                  // overlap next chunk copy
            stage_b_issue(BT16, bstage + ((ks + 1) & 1) * 32 * BSTRIDE, nbase, ks + 1);

        const _Float16* ap = arow + ks * 32;
        v8h alo = *(const v8h*)(ap + khalf);
        v8h ahi = *(const v8h*)(ap + 16 + khalf);
        v16h a = __builtin_shufflevector(alo, ahi, 0,1,2,3,4,5,6,7,8,9,10,11,12,13,14,15);

        const _Float16* bl = bstage + (ks & 1) * 32 * BSTRIDE + l16;
#pragma unroll
        for (int t = 0; t < 4; ++t) {
            v16h b;
#pragma unroll
            for (int s = 0; s < 8; ++s) {
                b[s]     = bl[(khalf + s) * BSTRIDE + t * 16];
                b[s + 8] = bl[(16 + khalf + s) * BSTRIDE + t * 16];
            }
            acc[t] = __builtin_amdgcn_wmma_f32_16x16x32_f16(
                false, a, false, b, (short)0, acc[t], false, false);
        }
    }
}

// C/D layout: lane -> n = l16 (col in tile), m = khalf + v for v8f element v.

// Pass 1: exp scores -> row/col sums
__global__ void gemm_sums_kernel(const _Float16* __restrict__ A16,
                                 const _Float16* __restrict__ BT16,
                                 float* __restrict__ rowsum,
                                 float* __restrict__ colsum) {
    __shared__ _Float16 bstage[2 * 32 * BSTRIDE];
    const int wave  = threadIdx.x >> 5;
    const int lane  = threadIdx.x & 31;
    const int l16   = lane & 15;
    const int khalf = (lane >> 4) << 3;
    const int mtile = blockIdx.x * 8 + wave;   // 64 blocks * 8 waves = 512 m-tiles
    const int nbase = blockIdx.y * 64;         // 128 n-strips

    v8f acc[4];
    wmma_strip(A16, BT16, bstage, mtile, nbase, acc);

    float rpart[8];
#pragma unroll
    for (int v = 0; v < 8; ++v) rpart[v] = 0.f;
#pragma unroll
    for (int t = 0; t < 4; ++t) {
        float csum = 0.f;
#pragma unroll
        for (int v = 0; v < 8; ++v) {
            float s = __expf(2.f * acc[t][v] - 2.f);
            csum += s;
            rpart[v] += s;
        }
        csum += __shfl_xor(csum, 16, 32);      // combine the two m-halves
        if (lane < 16) atomicAdd(&colsum[nbase + t * 16 + lane], csum);
    }
#pragma unroll
    for (int off = 1; off < 16; off <<= 1) {
#pragma unroll
        for (int v = 0; v < 8; ++v) rpart[v] += __shfl_xor(rpart[v], off, 32);
    }
    if (l16 == 0) {
#pragma unroll
        for (int v = 0; v < 8; ++v)
            atomicAdd(&rowsum[mtile * 16 + khalf + v], rpart[v]);
    }
}

// Pass 2: normalized scores -> float-bit-prefix histogram (LDS, then flush)
__global__ void gemm_hist_kernel(const _Float16* __restrict__ A16,
                                 const _Float16* __restrict__ BT16,
                                 const float* __restrict__ rowsum,
                                 const float* __restrict__ colsum,
                                 unsigned* __restrict__ hist) {
    __shared__ _Float16 bstage[2 * 32 * BSTRIDE];
    __shared__ unsigned lhist[HBINS];
    for (int i = threadIdx.x; i < HBINS; i += blockDim.x) lhist[i] = 0u;
    __syncthreads();

    const int wave  = threadIdx.x >> 5;
    const int lane  = threadIdx.x & 31;
    const int l16   = lane & 15;
    const int khalf = (lane >> 4) << 3;
    const int mtile = blockIdx.x * 8 + wave;
    const int nbase = blockIdx.y * 64;

    v8f acc[4];
    wmma_strip(A16, BT16, bstage, mtile, nbase, acc);

    float rinv[8];
#pragma unroll
    for (int v = 0; v < 8; ++v) rinv[v] = 1.f / rowsum[mtile * 16 + khalf + v];
#pragma unroll
    for (int t = 0; t < 4; ++t) {
        float cinv = 1.f / colsum[nbase + t * 16 + l16];
#pragma unroll
        for (int v = 0; v < 8; ++v) {
            float s  = __expf(2.f * acc[t][v] - 2.f);
            float ns = s * s * rinv[v] * cinv;
            unsigned bin = __float_as_uint(ns) >> 19;  // positive floats: bin < 4096
            atomicAdd(&lhist[bin], 1u);
        }
    }
    __syncthreads();
    for (int i = threadIdx.x; i < HBINS; i += blockDim.x) {
        unsigned h = lhist[i];
        if (h) atomicAdd(&hist[i], h);
    }
}

// threshold scan: highest bins down until >= KTOP elements
__global__ void scan_hist_kernel(const unsigned* __restrict__ hist,
                                 unsigned* __restrict__ thrbits) {
    if (threadIdx.x == 0 && blockIdx.x == 0) {
        unsigned cum = 0;
        int b = HBINS - 1;
        for (; b >= 0; --b) {
            cum += hist[b];
            if (cum >= KTOP) break;
        }
        if (b < 0) b = 0;
        *thrbits = (unsigned)b << 19;   // lower edge of threshold bin
    }
}

// Pass 3: collect candidates >= threshold
__global__ void gemm_collect_kernel(const _Float16* __restrict__ A16,
                                    const _Float16* __restrict__ BT16,
                                    const float* __restrict__ rowsum,
                                    const float* __restrict__ colsum,
                                    const unsigned* __restrict__ thrbits,
                                    unsigned* __restrict__ counter,
                                    uint2* __restrict__ cand) {
    __shared__ _Float16 bstage[2 * 32 * BSTRIDE];
    const int wave  = threadIdx.x >> 5;
    const int lane  = threadIdx.x & 31;
    const int l16   = lane & 15;
    const int khalf = (lane >> 4) << 3;
    const int mtile = blockIdx.x * 8 + wave;
    const int nbase = blockIdx.y * 64;
    const unsigned thr = *thrbits;

    v8f acc[4];
    wmma_strip(A16, BT16, bstage, mtile, nbase, acc);

    float rinv[8];
#pragma unroll
    for (int v = 0; v < 8; ++v) rinv[v] = 1.f / rowsum[mtile * 16 + khalf + v];
#pragma unroll
    for (int t = 0; t < 4; ++t) {
        int n = nbase + t * 16 + l16;
        float cinv = 1.f / colsum[n];
#pragma unroll
        for (int v = 0; v < 8; ++v) {
            float s  = __expf(2.f * acc[t][v] - 2.f);
            float ns = s * s * rinv[v] * cinv;
            unsigned bits = __float_as_uint(ns);
            if (bits >= thr) {
                unsigned slot = atomicAdd(counter, 1u);
                if (slot < CAP) {
                    int m = mtile * 16 + khalf + v;
                    cand[slot] = make_uint2(bits, (unsigned)(m * NPTS + n));
                }
            }
        }
    }
}

// final: bitonic sort candidates (bits desc, idx asc), take 512, NN-gate, emit
__global__ void finalize_kernel(const uint2* __restrict__ cand,
                                const unsigned* __restrict__ counter,
                                const int* __restrict__ ref_nn,
                                const int* __restrict__ src_nn,
                                const float* __restrict__ rowsum,
                                const float* __restrict__ colsum,
                                const float* __restrict__ ref_feats,
                                const float* __restrict__ src_feats,
                                float* __restrict__ out) {
    __shared__ unsigned long long skey[CAP];
    const int tid = threadIdx.x;
    const int NT = 512;
    unsigned cnt = *counter;
    if (cnt > CAP) cnt = CAP;
    for (unsigned i = tid; i < CAP; i += NT) {
        unsigned long long key = 0ull;
        if (i < cnt) {
            uint2 c = cand[i];
            key = ((unsigned long long)c.x << 32) | (unsigned)(~c.y); // desc bits, asc idx
        }
        skey[i] = key;
    }
    __syncthreads();
    for (unsigned k = 2; k <= CAP; k <<= 1) {
        for (unsigned j = k >> 1; j > 0; j >>= 1) {
            for (unsigned i = tid; i < CAP; i += NT) {
                unsigned p = i ^ j;
                if (p > i) {
                    unsigned long long a = skey[i], b = skey[p];
                    bool up = ((i & k) == 0);
                    if (up ? (a < b) : (a > b)) { skey[i] = b; skey[p] = a; }
                }
            }
            __syncthreads();
        }
    }
    if (tid < KTOP) {
        unsigned long long key = skey[tid];
        unsigned bits = (unsigned)(key >> 32);
        unsigned idx  = ~((unsigned)(key & 0xffffffffull));
        float o_ref = -1.f, o_src = -1.f, o_sc = 0.f;
        if (bits != 0u) {
            int rsel = (int)(idx >> 13);        // / 8192
            int ssel = (int)(idx & 8191u);      // % 8192
            int rn = ref_nn[rsel];
            int sn = src_nn[ssel];
            const float* fr = ref_feats + (size_t)rn * FDIM;
            const float* fs = src_feats + (size_t)sn * FDIM;
            float dot = 0.f;
            for (int kk = 0; kk < FDIM; ++kk) dot += fr[kk] * fs[kk];
            float s  = __expf(2.f * dot - 2.f);
            float ns = s * s / (rowsum[rn] * colsum[sn]);
            if (ns > (1.0f / (8192.f * 8192.f))) {
                o_ref = (float)rsel;
                o_src = (float)ssel;
                o_sc  = __uint_as_float(bits);
            }
        }
        out[tid]            = o_ref;
        out[KTOP + tid]     = o_src;
        out[2 * KTOP + tid] = o_sc;
    }
}

// ---------------------------------------------------------------- launcher

extern "C" void kernel_launch(void* const* d_in, const int* in_sizes, int n_in,
                              void* d_out, int out_size, void* d_ws, size_t ws_size,
                              hipStream_t stream) {
    (void)in_sizes; (void)n_in; (void)out_size; (void)ws_size;
    const float* ref_feats  = (const float*)d_in[0];
    const float* src_feats  = (const float*)d_in[1];
    const float* ref_points = (const float*)d_in[2];
    const float* src_points = (const float*)d_in[3];
    float* out = (float*)d_out;

    unsigned* wsu   = (unsigned*)d_ws;
    float*    rowsum = (float*)(wsu + WS_ROWSUM);
    float*    colsum = (float*)(wsu + WS_COLSUM);
    int*      ref_nn = (int*)(wsu + WS_REFNN);
    int*      src_nn = (int*)(wsu + WS_SRCNN);
    unsigned* hist   = wsu + WS_HIST;
    unsigned* counter = wsu + WS_CNT;
    unsigned* thrbits = wsu + WS_THR;
    uint2*    cand   = (uint2*)(wsu + WS_CAND);
    _Float16* ref16  = (_Float16*)(wsu + WS_REF16);
    _Float16* srcT16 = (_Float16*)(wsu + WS_SRCT16);

    // 0) zero accumulators/histogram/counter
    init_ws_kernel<<<(WS_ZERO_DWORDS + 255) / 256, 256, 0, stream>>>(wsu);
    // 1) precision conversion for WMMA operands
    cvt_f16_kernel<<<(NPTS * FDIM) / 256, 256, 0, stream>>>(ref_feats, ref16);
    cvt_f16_T_kernel<<<dim3(FDIM / 16, NPTS / 16), 256, 0, stream>>>(src_feats, srcT16);
    // 2) point nearest neighbors
    nn_kernel<<<NPTS / 256, 256, 0, stream>>>(ref_points, ref_nn);
    nn_kernel<<<NPTS / 256, 256, 0, stream>>>(src_points, src_nn);
    // 3) pass 1: WMMA GEMM (async-LDS staged B) -> exp -> row/col sums
    dim3 ggrid(512 / 8, NPTS / 64);
    gemm_sums_kernel<<<ggrid, 256, 0, stream>>>(ref16, srcT16, rowsum, colsum);
    // 4) pass 2: histogram of normalized-score float bits
    gemm_hist_kernel<<<ggrid, 256, 0, stream>>>(ref16, srcT16, rowsum, colsum, hist);
    // 5) threshold for top-512
    scan_hist_kernel<<<1, 32, 0, stream>>>(hist, thrbits);
    // 6) pass 3: collect candidates above threshold
    gemm_collect_kernel<<<ggrid, 256, 0, stream>>>(ref16, srcT16, rowsum, colsum,
                                                   thrbits, counter, cand);
    // 7) sort, select top-512, NN validity gate, emit outputs
    finalize_kernel<<<1, 512, 0, stream>>>(cand, counter, ref_nn, src_nn,
                                           rowsum, colsum, ref_feats, src_feats, out);
}
